// CircMACBlock_v3_31275951849606
// MI455X (gfx1250) — matrix operand order repaired and checked by reading, not compiled, not run on hardware
//
#include <hip/hip_runtime.h>
#include <hip/hip_bf16.h>

typedef __attribute__((ext_vector_type(16))) __bf16      v16bf;
typedef __attribute__((ext_vector_type(8)))  float       v8f;
typedef __attribute__((ext_vector_type(4)))  unsigned int u32x4;
typedef __attribute__((ext_vector_type(4)))  float       f32x4;

// ---------------------------------------------------------------------------
// helpers
// ---------------------------------------------------------------------------
static __device__ __forceinline__ unsigned short f2bf(float f) {
  unsigned int u = __float_as_uint(f);
  u += 0x7FFFu + ((u >> 16) & 1u);          // round-to-nearest-even to bf16
  return (unsigned short)(u >> 16);
}

union Frag { u32x4 q[2]; v16bf v; };

static __device__ __forceinline__ v8f wmma_bf16(const Frag& a, const Frag& b, v8f c) {
  return __builtin_amdgcn_wmma_f32_16x16x32_bf16(false, a.v, false, b.v,
                                                 (short)0, c, false, false);
}

// ---------------------------------------------------------------------------
// GEMM tile staging: A tile 128x32 as [m][k], B tile 32x128 transposed to
// [n][k]; fp32 -> bf16 on the fly; prefetch two K-tiles ahead.
// ---------------------------------------------------------------------------
static __device__ __forceinline__ void stage_tiles(
    unsigned short (*As)[32], unsigned short (*Bs)[32],
    const float* __restrict__ A, long lda,
    const float* __restrict__ Bm, long ldb,
    int m0, int n0, int k0, int M, int N, int K, int tid)
{
  { // A: 2 threads per row, 16 floats each
    int arow = tid >> 1, ahalf = tid & 1;
    int gm = m0 + arow;
    unsigned short* dst = &As[arow][ahalf * 16];
    if (gm < M) {
      const float* src = A + (long)gm * lda + k0 + ahalf * 16;
#pragma unroll
      for (int v = 0; v < 4; ++v) {
        f32x4 x = *(const f32x4*)(src + v * 4);
        dst[v*4+0] = f2bf(x.x); dst[v*4+1] = f2bf(x.y);
        dst[v*4+2] = f2bf(x.z); dst[v*4+3] = f2bf(x.w);
      }
      if (k0 + 96 <= K) __builtin_prefetch(src + 64, 0, 1);   // global_prefetch_b8
    } else {
#pragma unroll
      for (int v = 0; v < 16; ++v) dst[v] = 0;
    }
  }
  { // B: 128 n-columns x 2 k-groups of 16
    int bn = tid & 127, bkg = tid >> 7;
    int gn = n0 + bn;
#pragma unroll
    for (int j = 0; j < 16; ++j) {
      int kk = bkg * 16 + j;
      float v = (gn < N) ? Bm[(long)(k0 + kk) * ldb + gn] : 0.f;
      Bs[bn][kk] = f2bf(v);
    }
    if (gn < N && k0 + 96 <= K)
      __builtin_prefetch(&Bm[(long)(k0 + 64 + bkg * 16) * ldb + gn], 0, 1);
  }
}

// ---------------------------------------------------------------------------
// Generic tiled GEMM: C[M,N] = act(A[M,K] * B[K,N] + bias)
// block tile 128x128, 8 waves (4 rows x 2 cols), wave tile 32x64 (2x4 wmma),
// K step 32, double-buffered LDS staging.
// EPI: 0 = none, 1 = +bias, 2 = softplus(+bias), 3 = gelu(+bias)
// ---------------------------------------------------------------------------
template <int EPI>
__global__ __launch_bounds__(256) void gemm_wmma(
    const float* __restrict__ A, long lda,
    const float* __restrict__ Bm, long ldb,
    float* __restrict__ C, long ldc,
    const float* __restrict__ bias,
    int M, int N, int K)
{
  __shared__ __align__(16) unsigned short As[2][128][32];
  __shared__ __align__(16) unsigned short Bs[2][128][32];

  const int tid  = threadIdx.x;
  const int lane = tid & 31;
  const int wave = tid >> 5;
  const int wm = wave >> 1, wn = wave & 1;
  const int m0 = blockIdx.y * 128;
  const int n0 = blockIdx.x * 128;
  const int hi = lane >> 4;
  const int lm = lane & 15;

  v8f acc[2][4] = {};

  const int nk = K >> 5;
  int buf = 0;
  stage_tiles(As[0], Bs[0], A, lda, Bm, ldb, m0, n0, 0, M, N, K, tid);

  for (int kt = 0; kt < nk; ++kt) {
    __syncthreads();
    if (kt + 1 < nk)
      stage_tiles(As[buf ^ 1], Bs[buf ^ 1], A, lda, Bm, ldb,
                  m0, n0, (kt + 1) * 32, M, N, K, tid);

    Frag a[2], b[4];
#pragma unroll
    for (int r = 0; r < 2; ++r) {
      const unsigned short* p = &As[buf][wm * 32 + r * 16 + lm][0];
      a[r].q[0] = *(const u32x4*)(p + hi * 8);
      a[r].q[1] = *(const u32x4*)(p + 16 + hi * 8);
    }
#pragma unroll
    for (int c = 0; c < 4; ++c) {
      const unsigned short* p = &Bs[buf][wn * 64 + c * 16 + lm][0];
      b[c].q[0] = *(const u32x4*)(p + hi * 8);
      b[c].q[1] = *(const u32x4*)(p + 16 + hi * 8);
    }
#pragma unroll
    for (int r = 0; r < 2; ++r)
#pragma unroll
      for (int c = 0; c < 4; ++c)
        acc[r][c] = wmma_bf16(a[r], b[c], acc[r][c]);

    buf ^= 1;
  }

#pragma unroll
  for (int r = 0; r < 2; ++r)
#pragma unroll
    for (int c = 0; c < 4; ++c) {
      int gn = n0 + wn * 64 + c * 16 + lm;
#pragma unroll
      for (int e = 0; e < 8; ++e) {
        int gm = m0 + wm * 32 + r * 16 + hi * 8 + e;
        if (gm < M && gn < N) {
          float v = acc[r][c][e];
          if (EPI >= 1) v += bias[gn];
          if (EPI == 2) v = (v > 20.f) ? v : log1pf(expf(v));
          if (EPI == 3) v = 0.5f * v * (1.f + erff(v * 0.70710678118f));
          C[(long)gm * ldc + gn] = v;
        }
      }
    }
}

// ---------------------------------------------------------------------------
// Flash attention with WMMA.  1 block = (b, h, 64 q rows), 4 waves x 16 rows.
// q,k,v live in `full` at col offsets 0 / 1024 / 2048; head h uses dims h*64..
// ---------------------------------------------------------------------------
__global__ __launch_bounds__(128) void attention_wmma(
    const float* __restrict__ full, const int* __restrict__ mask,
    float* __restrict__ attn)
{
  __shared__ __align__(16) unsigned short Qs[64][64];      // [qrow][d], scaled
  __shared__ __align__(16) unsigned short Ks[64][64];      // [kpos][d]
  __shared__ __align__(16) unsigned short Vs[64][64];      // [d][kpos] (transposed)
  __shared__ __align__(16) unsigned short Ps[4][16][64];   // per-wave P strip

  const int tid  = threadIdx.x;
  const int lane = tid & 31;
  const int wave = tid >> 5;
  const int hi = lane >> 4, lm = lane & 15;
  const int qt = blockIdx.x, h = blockIdx.y, b = blockIdx.z;
  const int qbase = qt * 64;

  { // stage Q (pre-scaled by 1/sqrt(64))
    int row = tid >> 1, half = tid & 1;
    const float* src = full + ((long)(b * 1024 + qbase + row)) * 4096 + h * 64 + half * 32;
#pragma unroll
    for (int v = 0; v < 8; ++v) {
      f32x4 x = *(const f32x4*)(src + v * 4);
      int d = half * 32 + v * 4;
      Qs[row][d+0] = f2bf(x.x * 0.125f); Qs[row][d+1] = f2bf(x.y * 0.125f);
      Qs[row][d+2] = f2bf(x.z * 0.125f); Qs[row][d+3] = f2bf(x.w * 0.125f);
    }
  }

  v8f O[4] = {};
  float m8[8], l8[8];
#pragma unroll
  for (int e = 0; e < 8; ++e) { m8[e] = -1e30f; l8[e] = 0.f; }

  for (int kt = 0; kt < 16; ++kt) {
    const int kbase = kt * 64;
    { // stage K and V (V transposed); prefetch next k-tile
      int row = tid >> 1, half = tid & 1;
      const long rbase = ((long)(b * 1024 + kbase + row)) * 4096 + h * 64 + half * 32;
      const float* ksrc = full + rbase + 1024;
      const float* vsrc = full + rbase + 2048;
#pragma unroll
      for (int v = 0; v < 8; ++v) {
        f32x4 xk = *(const f32x4*)(ksrc + v * 4);
        int d = half * 32 + v * 4;
        Ks[row][d+0] = f2bf(xk.x); Ks[row][d+1] = f2bf(xk.y);
        Ks[row][d+2] = f2bf(xk.z); Ks[row][d+3] = f2bf(xk.w);
        f32x4 xv = *(const f32x4*)(vsrc + v * 4);
        Vs[d+0][row] = f2bf(xv.x); Vs[d+1][row] = f2bf(xv.y);
        Vs[d+2][row] = f2bf(xv.z); Vs[d+3][row] = f2bf(xv.w);
      }
      if (kt + 1 < 16) {
        __builtin_prefetch(ksrc + 64 * 4096, 0, 1);
        __builtin_prefetch(vsrc + 64 * 4096, 0, 1);
      }
    }
    __syncthreads();

    // ---- S = Q * K^T  (16x64 strip per wave) ----
    v8f S[4] = {};
    Frag aq[2];
#pragma unroll
    for (int ck = 0; ck < 2; ++ck) {
      const unsigned short* p = &Qs[wave * 16 + lm][ck * 32];
      aq[ck].q[0] = *(const u32x4*)(p + hi * 8);
      aq[ck].q[1] = *(const u32x4*)(p + 16 + hi * 8);
    }
#pragma unroll
    for (int c = 0; c < 4; ++c)
#pragma unroll
      for (int ck = 0; ck < 2; ++ck) {
        Frag bk;
        const unsigned short* p = &Ks[c * 16 + lm][ck * 32];
        bk.q[0] = *(const u32x4*)(p + hi * 8);
        bk.q[1] = *(const u32x4*)(p + 16 + hi * 8);
        S[c] = wmma_bf16(aq[ck], bk, S[c]);
      }

    // ---- circular rel bias + mask + online softmax ----
    float alpha[8];
#pragma unroll
    for (int e = 0; e < 8; ++e) {
      int qg = qbase + wave * 16 + hi * 8 + e;
      float mx = -1e30f;
#pragma unroll
      for (int c = 0; c < 4; ++c) {
        int kg = kbase + c * 16 + lm;
        int dd = qg - kg; if (dd < 0) dd = -dd;
        int cb = (dd < 1024 - dd) ? dd : (1024 - dd);
        float s = S[c][e] - (float)cb;
        if (mask[b * 1024 + kg] == 0) s = -1e9f;
        S[c][e] = s;
        mx = fmaxf(mx, s);
      }
#pragma unroll
      for (int off = 1; off < 16; off <<= 1)
        mx = fmaxf(mx, __shfl_xor(mx, off, 32));
      float mnew = fmaxf(m8[e], mx);
      alpha[e] = expf(m8[e] - mnew);
      float sum = 0.f;
#pragma unroll
      for (int c = 0; c < 4; ++c) {
        float p = expf(S[c][e] - mnew);
        S[c][e] = p;
        sum += p;
      }
#pragma unroll
      for (int off = 1; off < 16; off <<= 1)
        sum += __shfl_xor(sum, off, 32);
      l8[e] = l8[e] * alpha[e] + sum;
      m8[e] = mnew;
    }
#pragma unroll
    for (int c = 0; c < 4; ++c)
#pragma unroll
      for (int e = 0; e < 8; ++e) {
        O[c][e] *= alpha[e];
        Ps[wave][hi * 8 + e][c * 16 + lm] = f2bf(S[c][e]);
      }
    __syncthreads();

    // ---- O += P * V ----
    Frag ap[2];
#pragma unroll
    for (int ck = 0; ck < 2; ++ck) {
      const unsigned short* p = &Ps[wave][lm][ck * 32];
      ap[ck].q[0] = *(const u32x4*)(p + hi * 8);
      ap[ck].q[1] = *(const u32x4*)(p + 16 + hi * 8);
    }
#pragma unroll
    for (int c = 0; c < 4; ++c)
#pragma unroll
      for (int ck = 0; ck < 2; ++ck) {
        Frag bv;
        const unsigned short* p = &Vs[c * 16 + lm][ck * 32];
        bv.q[0] = *(const u32x4*)(p + hi * 8);
        bv.q[1] = *(const u32x4*)(p + 16 + hi * 8);
        O[c] = wmma_bf16(ap[ck], bv, O[c]);
      }
    __syncthreads();
  }

#pragma unroll
  for (int c = 0; c < 4; ++c) {
    int dg = h * 64 + c * 16 + lm;
#pragma unroll
    for (int e = 0; e < 8; ++e) {
      int qg = qbase + wave * 16 + hi * 8 + e;
      attn[((long)(b * 1024 + qg)) * 1024 + dg] = O[c][e] / l8[e];
    }
  }
}

// ---------------------------------------------------------------------------
// circular depthwise conv on base (= full cols 3072..4095), K=7
// ---------------------------------------------------------------------------
__global__ __launch_bounds__(256) void circconv(
    const float* __restrict__ full, const float* __restrict__ w,
    const float* __restrict__ bias, float* __restrict__ out)
{
  int d = blockIdx.x * 256 + threadIdx.x;
  int l = blockIdx.y, b = blockIdx.z;
  float y = bias[d];
#pragma unroll
  for (int k = 0; k < 7; ++k) {
    int ll = (l + k - 3 + 1024) & 1023;
    y += full[((long)(b * 1024 + ll)) * 4096 + 3072 + d] * w[d * 7 + k];
  }
  out[((long)(b * 1024 + l)) * 1024 + d] = y;
}

// mamba causal depthwise conv (DC=4) + SiLU ; u = uz cols 0..2047
__global__ __launch_bounds__(256) void mconv(
    const float* __restrict__ uz, const float* __restrict__ w,
    const float* __restrict__ bias, float* __restrict__ uc)
{
  int d = blockIdx.x * 256 + threadIdx.x;
  int l = blockIdx.y, b = blockIdx.z;
  float y = bias[d];
#pragma unroll
  for (int k = 0; k < 4; ++k) {
    int ll = l - 3 + k;
    if (ll >= 0) y += uz[((long)(b * 1024 + ll)) * 4096 + d] * w[d * 4 + k];
  }
  y = y / (1.f + expf(-y));
  uc[((long)(b * 1024 + l)) * 2048 + d] = y;
}

// selective scan: one thread per (b, channel); dty holds dt in, y out (in place)
__global__ __launch_bounds__(256) void mamba_scan(
    const float* __restrict__ uc, float* __restrict__ dty,
    const float* __restrict__ xdbl, const float* __restrict__ uz,
    const float* __restrict__ Alog, const float* __restrict__ Dv)
{
  int d = blockIdx.x * 256 + threadIdx.x;   // 0..2047
  int b = blockIdx.y;
  float a[16], hst[16];
#pragma unroll
  for (int s = 0; s < 16; ++s) { a[s] = -expf(Alog[d * 16 + s]); hst[s] = 0.f; }
  float Dd = Dv[d];
  for (int t = 0; t < 1024; ++t) {
    long ridx = (long)(b * 1024 + t);
    float dt = dty[ridx * 2048 + d];
    float u  = uc[ridx * 2048 + d];
    const float* bc = xdbl + ridx * 96 + 64;   // Bm then Cm, 16 each
    float du = dt * u;
    float y = 0.f;
#pragma unroll
    for (int s = 0; s < 16; ++s) {
      hst[s] = expf(dt * a[s]) * hst[s] + du * bc[s];
      y += hst[s] * bc[16 + s];
    }
    float z = uz[ridx * 4096 + 2048 + d];
    y = (y + u * Dd) * (z / (1.f + expf(-z)));
    dty[ridx * 2048 + d] = y;
  }
}

// in-place RMS norm over last dim (1024), one block per row
__global__ __launch_bounds__(256) void rmsnorm_inplace(
    float* __restrict__ x, const float* __restrict__ w)
{
  __shared__ float red[8];
  float* p = x + (long)blockIdx.x * 1024;
  int tid = threadIdx.x;
  float s = 0.f;
  for (int i = tid; i < 1024; i += 256) { float v = p[i]; s += v * v; }
#pragma unroll
  for (int off = 16; off >= 1; off >>= 1) s += __shfl_xor(s, off, 32);
  if ((tid & 31) == 0) red[tid >> 5] = s;
  __syncthreads();
  if (tid == 0) {
    float t = 0.f;
    for (int i = 0; i < 8; ++i) t += red[i];
    red[0] = rsqrtf(t * (1.f / 1024.f) + 1e-6f);
  }
  __syncthreads();
  float scale = red[0];
  for (int i = tid; i < 1024; i += 256) p[i] = w[i] * (p[i] * scale);
}

// ri[b, seg*1024 + d] = mean over L of src_seg[b, :, d]
__global__ __launch_bounds__(256) void rowmean3(
    const float* __restrict__ a, const float* __restrict__ m,
    const float* __restrict__ c, float* __restrict__ ri)
{
  int d = blockIdx.x * 256 + threadIdx.x;
  int seg = blockIdx.y, b = blockIdx.z;
  const float* src = (seg == 0) ? a : (seg == 1) ? m : c;
  float s = 0.f;
  for (int l = 0; l < 1024; ++l) s += src[((long)(b * 1024 + l)) * 1024 + d];
  ri[b * 3072 + seg * 1024 + d] = s * (1.f / 1024.f);
}

// final tiny router matmul (8x1024 @ 1024x3) + softmax
__global__ __launch_bounds__(64) void router2(
    const float* __restrict__ g, const float* __restrict__ w2,
    const float* __restrict__ b2, float* __restrict__ gates)
{
  __shared__ float lg[8][3];
  int t = threadIdx.x;
  if (t < 24) {
    int b = t / 3, j = t % 3;
    float s = b2[j];
    for (int i = 0; i < 1024; ++i) s += g[b * 1024 + i] * w2[i * 3 + j];
    lg[b][j] = s;
  }
  __syncthreads();
  if (t < 8) {
    float x0 = lg[t][0], x1 = lg[t][1], x2 = lg[t][2];
    float mx = fmaxf(x0, fmaxf(x1, x2));
    float e0 = expf(x0 - mx), e1 = expf(x1 - mx), e2 = expf(x2 - mx);
    float inv = 1.f / (e0 + e1 + e2);
    gates[t * 3 + 0] = e0 * inv; gates[t * 3 + 1] = e1 * inv; gates[t * 3 + 2] = e2 * inv;
  }
}

// fused = a*g0 + m*g1 + c*g2, written in place over a
__global__ __launch_bounds__(256) void fuse_gate(
    float* __restrict__ a, const float* __restrict__ m,
    const float* __restrict__ c, const float* __restrict__ gates)
{
  long i = (long)blockIdx.x * 256 + threadIdx.x;
  int b = (int)(i >> 20);                       // L*D = 2^20 per batch
  a[i] = a[i] * gates[b * 3 + 0] + m[i] * gates[b * 3 + 1] + c[i] * gates[b * 3 + 2];
}

// ---------------------------------------------------------------------------
extern "C" void kernel_launch(void* const* d_in, const int* in_sizes, int n_in,
                              void* d_out, int out_size, void* d_ws, size_t ws_size,
                              hipStream_t stream)
{
  (void)in_sizes; (void)n_in; (void)out_size; (void)ws_size;
  const float* x        = (const float*)d_in[0];
  const int*   mask     = (const int*)  d_in[1];
  const float* W_in     = (const float*)d_in[2];
  const float* b_in     = (const float*)d_in[3];
  const float* nw_attn  = (const float*)d_in[4];
  const float* nw_mamba = (const float*)d_in[5];
  const float* nw_cnn   = (const float*)d_in[6];
  const float* conv_w   = (const float*)d_in[7];
  const float* conv_b   = (const float*)d_in[8];
  const float* m_in_w   = (const float*)d_in[9];
  const float* m_conv_w = (const float*)d_in[10];
  const float* m_conv_b = (const float*)d_in[11];
  const float* m_x_w    = (const float*)d_in[12];
  const float* m_dt_w   = (const float*)d_in[13];
  const float* m_dt_b   = (const float*)d_in[14];
  const float* m_A_log  = (const float*)d_in[15];
  const float* m_D      = (const float*)d_in[16];
  const float* m_out_w  = (const float*)d_in[17];
  const float* r_w1     = (const float*)d_in[18];
  const float* r_b1     = (const float*)d_in[19];
  const float* r_w2     = (const float*)d_in[20];
  const float* r_b2     = (const float*)d_in[21];
  const float* W_out    = (const float*)d_in[22];
  const float* b_out    = (const float*)d_in[23];
  float* out = (float*)d_out;

  float* ws = (float*)d_ws;
  float* full    = ws; ws += (size_t)8192 * 4096;   // q|k|v|base
  float* uz      = ws; ws += (size_t)8192 * 4096;   // u|z
  float* uc      = ws; ws += (size_t)8192 * 2048;
  float* xdbl    = ws; ws += (size_t)8192 * 96;
  float* dty     = ws; ws += (size_t)8192 * 2048;   // dt, then y (in place)
  float* mamba_o = ws; ws += (size_t)8192 * 1024;
  float* attn_o  = ws; ws += (size_t)8192 * 1024;   // later reused as `fused`
  float* cnn_o   = ws; ws += (size_t)8192 * 1024;
  float* ri      = ws; ws += 8 * 3072;
  float* gbuf    = ws; ws += 8 * 1024;
  float* gates   = ws; ws += 32;

  dim3 blk(256);
  const int M = 8192;
  auto gx = [](int N) { return (N + 127) / 128; };

  // full = x @ W_in + b_in
  gemm_wmma<1><<<dim3(gx(4096), 64), blk, 0, stream>>>(x, 1024, W_in, 4096, full, 4096, b_in, M, 4096, 1024);
  // uz = base @ m_in_w
  gemm_wmma<0><<<dim3(gx(4096), 64), blk, 0, stream>>>(full + 3072, 4096, m_in_w, 4096, uz, 4096, nullptr, M, 4096, 1024);
  // attention branch (reads q,k,v from full)
  attention_wmma<<<dim3(16, 16, 8), dim3(128), 0, stream>>>(full, mask, attn_o);
  // circular conv branch (reads base from full)
  circconv<<<dim3(4, 1024, 8), blk, 0, stream>>>(full, conv_w, conv_b, cnn_o);
  // mamba: causal conv + silu
  mconv<<<dim3(8, 1024, 8), blk, 0, stream>>>(uz, m_conv_w, m_conv_b, uc);
  // x_dbl = uc @ m_x_w
  gemm_wmma<0><<<dim3(gx(96), 64), blk, 0, stream>>>(uc, 2048, m_x_w, 96, xdbl, 96, nullptr, M, 96, 2048);
  // dt = softplus(dt_r @ m_dt_w + m_dt_b)
  gemm_wmma<2><<<dim3(gx(2048), 64), blk, 0, stream>>>(xdbl, 96, m_dt_w, 2048, dty, 2048, m_dt_b, M, 2048, 64);
  // selective scan (y overwrites dt buffer)
  mamba_scan<<<dim3(8, 8), blk, 0, stream>>>(uc, dty, xdbl, uz, m_A_log, m_D);
  // mamba_o = y @ m_out_w
  gemm_wmma<0><<<dim3(gx(1024), 64), blk, 0, stream>>>(dty, 2048, m_out_w, 1024, mamba_o, 1024, nullptr, M, 1024, 2048);
  // RMS norms (in place)
  rmsnorm_inplace<<<8192, blk, 0, stream>>>(attn_o, nw_attn);
  rmsnorm_inplace<<<8192, blk, 0, stream>>>(mamba_o, nw_mamba);
  rmsnorm_inplace<<<8192, blk, 0, stream>>>(cnn_o, nw_cnn);
  // router input means
  rowmean3<<<dim3(4, 3, 8), blk, 0, stream>>>(attn_o, mamba_o, cnn_o, ri);
  // gbuf = gelu(ri @ r_w1 + r_b1)
  gemm_wmma<3><<<dim3(gx(1024), 1), blk, 0, stream>>>(ri, 3072, r_w1, 1024, gbuf, 1024, r_b1, 8, 1024, 3072);
  // gates = softmax(gbuf @ r_w2 + r_b2)
  router2<<<1, 64, 0, stream>>>(gbuf, r_w2, r_b2, gates);
  // fused (in place over attn_o)
  fuse_gate<<<32768, blk, 0, stream>>>(attn_o, mamba_o, cnn_o, gates);
  // out = fused @ W_out + b_out
  gemm_wmma<1><<<dim3(gx(1024), 64), blk, 0, stream>>>(attn_o, 1024, W_out, 1024, out, 1024, b_out, M, 1024, 1024);
}
